// GroupedQueryAttention_1932735283554
// MI455X (gfx1250) — compile-verified, matrix operand order checked by reading, not run on hardware
//
#include <hip/hip_runtime.h>
#include <hip/hip_bf16.h>
#include <math.h>

// Problem constants (from reference)
#define B_   2
#define S_   2048
#define HID_ 2048
#define NH_  32
#define G_   8
#define HG_  4
#define D_   64
#define GD_  512   // G_ * D_

typedef __bf16 bf16_t;
typedef __attribute__((ext_vector_type(16))) __bf16 v16bf;
typedef __attribute__((ext_vector_type(8)))  __bf16 v8bf;
typedef __attribute__((ext_vector_type(8)))  float  v8f;
typedef __attribute__((ext_vector_type(4)))  float  v4f;

__device__ __forceinline__ bf16_t f2bf(float f) {
  union { float f; unsigned u; } a; a.f = f;
  unsigned r = a.u + 0x7FFFu + ((a.u >> 16) & 1u);   // round-to-nearest-even
  union { unsigned short s; bf16_t b; } o; o.s = (unsigned short)(r >> 16);
  return o.b;
}

__device__ __forceinline__ v8f wmma_bf16(v16bf a, v16bf b, v8f c) {
  // D = A(16x32) x B(32x16) + C, f32 accumulate
  return __builtin_amdgcn_wmma_f32_16x16x32_bf16(false, a, false, b, (short)0, c,
                                                 false, false);
}

// Load one 16x32 bf16 fragment (A layout; also Bt rows = B columns).
// Lane l: row = l%16, K chunk = (l/16)*8 + {0..7} and +16.  Two b128 loads.
__device__ __forceinline__ v16bf load_frag(const bf16_t* base, int ld) {
  const int lane = threadIdx.x & 31;
  const int r  = lane & 15;
  const int ks = (lane >> 4) << 3;
  const bf16_t* p = base + (size_t)r * ld + ks;
  v8bf lo = *reinterpret_cast<const v8bf*>(p);
  v8bf hi = *reinterpret_cast<const v8bf*>(p + 16);
  v16bf f;
#pragma unroll
  for (int i = 0; i < 8; ++i) { f[i] = lo[i]; f[i + 8] = hi[i]; }
  return f;
}

// ---------------------------------------------------------------- convert
__global__ void cvt_f32_bf16(const float* __restrict__ in,
                             bf16_t* __restrict__ out, int n4) {
  int i = blockIdx.x * blockDim.x + threadIdx.x;
  if (i < n4) {
    v4f v = reinterpret_cast<const v4f*>(in)[i];
    union { bf16_t b[4]; uint2 u; } t;
#pragma unroll
    for (int j = 0; j < 4; ++j) t.b[j] = f2bf(v[j]);
    reinterpret_cast<uint2*>(out)[i] = t.u;
  }
}

// ---------------------------------------------------------------- GEMM
// C[m,n] = sum_k A[m,k] * Bt[n,k] + bias[n]
// mode 0: store bf16 row-major (M x N)
// mode 1: store f32  row-major (M x N)
// mode 2: store bf16 transposed as V^T: out[((b*G+g)*D + d)*S + t]
// Block: 256 threads = 8 waves (2 M x 4 N), wave tile 32M x 64N.
// K must be a multiple of 64 (double-buffered, ping-pong fragment sets).
__global__ __launch_bounds__(256)
void gemm_bf16(const bf16_t* __restrict__ A, const bf16_t* __restrict__ Bt,
               const float* __restrict__ bias, void* __restrict__ out,
               int M, int N, int K, int mode) {
  const int wave = threadIdx.x >> 5;
  const int lane = threadIdx.x & 31;
  const int wm = wave >> 2;                       // 0..1
  const int wn = wave & 3;                        // 0..3
  const int row0 = blockIdx.x * 64 + wm * 32;
  const int col0 = blockIdx.y * 256 + wn * 64;

  const v8f vzero = {0.f, 0.f, 0.f, 0.f, 0.f, 0.f, 0.f, 0.f};
  v8f acc[2][4];
#pragma unroll
  for (int i = 0; i < 2; ++i)
#pragma unroll
    for (int j = 0; j < 4; ++j) acc[i][j] = vzero;

  auto load_set = [&](int k0, v16bf a[2], v16bf b[4]) {
    a[0] = load_frag(A + (size_t)row0 * K + k0, K);
    a[1] = load_frag(A + (size_t)(row0 + 16) * K + k0, K);
#pragma unroll
    for (int j = 0; j < 4; ++j)
      b[j] = load_frag(Bt + (size_t)(col0 + 16 * j) * K + k0, K);
  };
  auto mma_set = [&](const v16bf a[2], const v16bf b[4]) {
#pragma unroll
    for (int j = 0; j < 4; ++j) {
      acc[0][j] = wmma_bf16(a[0], b[j], acc[0][j]);
      acc[1][j] = wmma_bf16(a[1], b[j], acc[1][j]);
    }
  };

  v16bf aP[2], bP[4], aQ[2], bQ[4];
  load_set(0, aP, bP);
  for (int k0 = 0; k0 < K; k0 += 64) {
    // branchless speculative prefetch two K-steps ahead (gfx1250 global_prefetch)
    __builtin_prefetch(A  + (size_t)row0 * K + k0 + 128, 0, 1);
    __builtin_prefetch(Bt + (size_t)col0 * K + k0 + 128, 0, 1);
    load_set(k0 + 32, aQ, bQ);                    // loads overlap these WMMAs:
    mma_set(aP, bP);
    if (k0 + 64 < K) load_set(k0 + 64, aP, bP);   // and these:
    mma_set(aQ, bQ);
  }

  // C layout: lane holds column n=lane%16, rows mb..mb+7 with mb = (lane/16)*8
  const int n  = lane & 15;
  const int mb = (lane >> 4) << 3;
#pragma unroll
  for (int i = 0; i < 2; ++i) {
#pragma unroll
    for (int j = 0; j < 4; ++j) {
      const int c0 = col0 + j * 16 + n;
      const float bi = bias[c0];
      const int r0 = row0 + i * 16 + mb;
      if (mode == 0) {
        bf16_t* O = (bf16_t*)out;
#pragma unroll
        for (int r = 0; r < 8; ++r)
          O[(size_t)(r0 + r) * N + c0] = f2bf(acc[i][j][r] + bi);
      } else if (mode == 1) {
        float* O = (float*)out;
#pragma unroll
        for (int r = 0; r < 8; ++r)
          O[(size_t)(r0 + r) * N + c0] = acc[i][j][r] + bi;
      } else {
        bf16_t* O = (bf16_t*)out;                 // V^T : (b, g, d, t)
        const int g = c0 >> 6, d = c0 & 63;
#pragma unroll
        for (int r = 0; r < 8; ++r) {
          const int m = r0 + r;
          const int bb = m / S_, t = m % S_;
          O[(((size_t)bb * G_ + g) * D_ + d) * S_ + t] = f2bf(acc[i][j][r] + bi);
        }
      }
    }
  }
}

// ---------------------------------------------------------------- flash attention
// grid.x = B*NH*(S/128); 256 threads = 8 waves; each wave: 16 query rows.
__global__ __launch_bounds__(256)
void flash_attn(const bf16_t* __restrict__ Q, const bf16_t* __restrict__ Kt,
                const bf16_t* __restrict__ Vt, const int* __restrict__ mask,
                bf16_t* __restrict__ Out) {
  __shared__ __align__(16) bf16_t pbuf[8 * 16 * 32];  // per-wave P staging

  const int wave = threadIdx.x >> 5;
  const int lane = threadIdx.x & 31;
  const int n  = lane & 15;
  const int mb = (lane >> 4) << 3;

  const int qpb  = S_ / 128;                     // q-blocks per head
  const int head = blockIdx.x / qpb;
  const int qblk = blockIdx.x % qpb;
  const int b = head / NH_;
  const int h = head % NH_;
  const int g = h / HG_;
  const int s0 = qblk * 128 + wave * 16;

  const bf16_t* Qp = Q  + ((size_t)(b * S_ + s0)) * HID_ + h * D_;
  const bf16_t* Kp = Kt + ((size_t)b * S_) * GD_ + g * D_;
  const bf16_t* Vp = Vt + ((size_t)(b * G_ + g) * D_) * S_;
  const int* mrow = mask + b * S_;
  bf16_t* pb = pbuf + wave * (16 * 32);

  v16bf qf0 = load_frag(Qp, HID_);               // d = 0..31
  v16bf qf1 = load_frag(Qp + 32, HID_);          // d = 32..63

  const v8f vzero = {0.f, 0.f, 0.f, 0.f, 0.f, 0.f, 0.f, 0.f};
  v8f o[4];
#pragma unroll
  for (int dt = 0; dt < 4; ++dt) o[dt] = vzero;
  float rmax[8], rsum[8];
#pragma unroll
  for (int r = 0; r < 8; ++r) { rmax[r] = -1e30f; rsum[r] = 0.f; }

  const float scale = 0.125f;                    // 1/sqrt(D)

  for (int t0 = 0; t0 < S_; t0 += 32) {
    // speculative prefetch of next key/value tiles (branchless)
    __builtin_prefetch(Kp + (size_t)(t0 + 32) * GD_, 0, 1);
    __builtin_prefetch(Vp + t0 + 32, 0, 1);

    // scores: Q(16xD) x K^T(Dx32)  -> two 16x16 f32 tiles
    v16bf k00 = load_frag(Kp + (size_t)t0 * GD_, GD_);
    v16bf k01 = load_frag(Kp + (size_t)t0 * GD_ + 32, GD_);
    v16bf k10 = load_frag(Kp + (size_t)(t0 + 16) * GD_, GD_);
    v16bf k11 = load_frag(Kp + (size_t)(t0 + 16) * GD_ + 32, GD_);
    v8f sc0 = vzero, sc1 = vzero;
    sc0 = wmma_bf16(qf0, k00, sc0);
    sc0 = wmma_bf16(qf1, k01, sc0);
    sc1 = wmma_bf16(qf0, k10, sc1);
    sc1 = wmma_bf16(qf1, k11, sc1);

    const float madd0 = mrow[t0 + n]      ? 0.f : -1e30f;
    const float madd1 = mrow[t0 + 16 + n] ? 0.f : -1e30f;

    float p0[8], p1[8], alpha[8];
#pragma unroll
    for (int r = 0; r < 8; ++r) {
      p0[r] = sc0[r] * scale + madd0;
      p1[r] = sc1[r] * scale + madd1;
    }
    // online softmax: rows are replicated across each 16-lane half
#pragma unroll
    for (int r = 0; r < 8; ++r) {
      float tm = fmaxf(p0[r], p1[r]);
      for (int off = 1; off < 16; off <<= 1)
        tm = fmaxf(tm, __shfl_xor(tm, off, 32));
      const float nm = fmaxf(rmax[r], tm);
      alpha[r] = __expf(rmax[r] - nm);
      rmax[r] = nm;
      p0[r] = __expf(p0[r] - nm);
      p1[r] = __expf(p1[r] - nm);
      float ts = p0[r] + p1[r];
      for (int off = 1; off < 16; off <<= 1)
        ts += __shfl_xor(ts, off, 32);
      rsum[r] = rsum[r] * alpha[r] + ts;
    }
    // rescale running output
#pragma unroll
    for (int dt = 0; dt < 4; ++dt)
#pragma unroll
      for (int r = 0; r < 8; ++r) o[dt][r] *= alpha[r];

    // C-layout -> A-fragment layout via per-wave LDS (16x32 bf16 row-major)
#pragma unroll
    for (int r = 0; r < 8; ++r) {
      pb[(mb + r) * 32 + n]      = f2bf(p0[r]);
      pb[(mb + r) * 32 + 16 + n] = f2bf(p1[r]);
    }
    v16bf pf = load_frag(pb, 32);

    // O += P(16x32) x V(32x64); V stored transposed (d-major rows over t)
#pragma unroll
    for (int dt = 0; dt < 4; ++dt) {
      v16bf vf = load_frag(Vp + (size_t)(dt * 16) * S_ + t0, S_);
      o[dt] = wmma_bf16(pf, vf, o[dt]);
    }
  }

  bf16_t* Op = Out + ((size_t)(b * S_ + s0)) * HID_ + h * D_;
#pragma unroll
  for (int dt = 0; dt < 4; ++dt)
#pragma unroll
    for (int r = 0; r < 8; ++r) {
      const float inv = 1.f / rsum[r];
      Op[(size_t)(mb + r) * HID_ + dt * 16 + n] = f2bf(o[dt][r] * inv);
    }
}

// ---------------------------------------------------------------- launch
extern "C" void kernel_launch(void* const* d_in, const int* in_sizes, int n_in,
                              void* d_out, int out_size, void* d_ws, size_t ws_size,
                              hipStream_t stream) {
  (void)in_sizes; (void)n_in; (void)out_size; (void)ws_size;
  const float* hs  = (const float*)d_in[0];
  const int*  mask = (const int*) d_in[1];
  const float* Wq  = (const float*)d_in[2];
  const float* bq  = (const float*)d_in[3];
  const float* Wk  = (const float*)d_in[4];
  const float* bk  = (const float*)d_in[5];
  const float* Wv  = (const float*)d_in[6];
  const float* bv  = (const float*)d_in[7];
  const float* Wo  = (const float*)d_in[8];
  const float* bo  = (const float*)d_in[9];
  float* out = (float*)d_out;

  char* ws = (char*)d_ws;
  size_t off = 0;
  auto alloc = [&](size_t bytes) -> char* {
    char* p = ws + off;
    off += (bytes + 255) & ~(size_t)255;
    return p;
  };
  bf16_t* hsb = (bf16_t*)alloc((size_t)B_ * S_ * HID_ * 2);
  bf16_t* qb  = (bf16_t*)alloc((size_t)B_ * S_ * HID_ * 2);
  bf16_t* kb  = (bf16_t*)alloc((size_t)B_ * S_ * GD_ * 2);
  bf16_t* vtb = (bf16_t*)alloc((size_t)B_ * S_ * GD_ * 2);
  bf16_t* ab  = (bf16_t*)alloc((size_t)B_ * S_ * HID_ * 2);
  bf16_t* wqb = (bf16_t*)alloc((size_t)HID_ * HID_ * 2);
  bf16_t* wkb = (bf16_t*)alloc((size_t)GD_ * HID_ * 2);
  bf16_t* wvb = (bf16_t*)alloc((size_t)GD_ * HID_ * 2);
  bf16_t* wob = (bf16_t*)alloc((size_t)HID_ * HID_ * 2);

  auto cvt = [&](const float* src, bf16_t* dst, size_t nelem) {
    int n4 = (int)(nelem / 4);
    cvt_f32_bf16<<<(n4 + 255) / 256, 256, 0, stream>>>(src, dst, n4);
  };
  cvt(hs, hsb, (size_t)B_ * S_ * HID_);
  cvt(Wq, wqb, (size_t)HID_ * HID_);
  cvt(Wk, wkb, (size_t)GD_ * HID_);
  cvt(Wv, wvb, (size_t)GD_ * HID_);
  cvt(Wo, wob, (size_t)HID_ * HID_);

  const int M = B_ * S_;
  // Q = hs @ Wq^T + bq   (bf16 out)
  gemm_bf16<<<dim3(M / 64, HID_ / 256), 256, 0, stream>>>(hsb, wqb, bq, qb, M, HID_, HID_, 0);
  // K = hs @ Wk^T + bk   (bf16 out)
  gemm_bf16<<<dim3(M / 64, GD_ / 256), 256, 0, stream>>>(hsb, wkb, bk, kb, M, GD_, HID_, 0);
  // V = hs @ Wv^T + bv   (bf16 out, stored transposed per (b,g): D x S)
  gemm_bf16<<<dim3(M / 64, GD_ / 256), 256, 0, stream>>>(hsb, wvb, bv, vtb, M, GD_, HID_, 2);
  // flash attention -> ab (bf16, row-major B*S x HID)
  flash_attn<<<B_ * NH_ * (S_ / 128), 256, 0, stream>>>(qb, kb, vtb, mask, ab);
  // out = ab @ Wo^T + bo (f32 out)
  gemm_bf16<<<dim3(M / 64, HID_ / 256), 256, 0, stream>>>(ab, wob, bo, out, M, HID_, HID_, 1);
}
